// _TransformerBlockQuantum_65481071396274
// MI455X (gfx1250) — matrix operand order repaired and checked by reading, not compiled, not run on hardware
//
#include <hip/hip_runtime.h>
#include <hip/hip_bf16.h>
#include <stdint.h>

typedef __attribute__((ext_vector_type(16))) _Float16 v16h;
typedef __attribute__((ext_vector_type(8)))  float    v8f;

union Frag {
    v16h     v;
    uint32_t u[8];
    _Float16 h[16];
};

#define WMMA_F16(A, B, C) \
    __builtin_amdgcn_wmma_f32_16x16x32_f16(false, (A), false, (B), (short)0, (C), false, false)

// Async copy: 16 bytes global -> LDS per lane, tracked by ASYNCcnt (CDNA5).
__device__ __forceinline__ void async_copy_b128(uint32_t lds_addr, uint64_t gaddr) {
    asm volatile("global_load_async_to_lds_b128 %0, %1, off"
                 :: "v"(lds_addr), "v"(gaddr)
                 : "memory");
}
__device__ __forceinline__ void wait_async0() {
    asm volatile("s_wait_asynccnt 0x0" ::: "memory");
}

// ---------------------------------------------------------------------------
// f32 -> f16 conversion
// ---------------------------------------------------------------------------
__global__ __launch_bounds__(256) void cvt_f16_kernel(const float* __restrict__ in,
                                                      _Float16* __restrict__ out, int n) {
    int i = blockIdx.x * 256 + threadIdx.x;
    if (i < n) out[i] = (_Float16)in[i];
}

// ---------------------------------------------------------------------------
// GEMM: C[M,N] = A[M,K] @ W[N,K]^T   (A,W f16; accum f32; OUT = f16 or f32)
// Workgroup = 4 waves (128 threads), block tile 64(M) x 64(N), K-step 32.
// A and W tiles staged in LDS via global_load_async_to_lds_b128; the W tile
// is reused by all four waves (each wave owns a 16-row strip of M).
// A-fragment layout (16-bit A 16x32): lane = M (mod 16), elems 2v,2v+1 hold
//   K = 16*(v/4) + 8*(lane/16) + 2*(v%4) (+1).  B-fragment: lane%16 = N, elem = K.
// ---------------------------------------------------------------------------
#define TILE_M 64
#define TILE_N 64
#define TILE_K 32

template <typename OUT>
__global__ __launch_bounds__(128) void gemm_kernel(const _Float16* __restrict__ A,
                                                   const _Float16* __restrict__ W,
                                                   OUT* __restrict__ C,
                                                   int M, int N, int K) {
    __shared__ _Float16 sA[TILE_M * TILE_K];  // [m_local][k_local]
    __shared__ _Float16 sW[TILE_N * TILE_K];  // [n_local][k_local]

    const int t    = threadIdx.x;   // 0..127
    const int lane = t & 31;
    const int wave = t >> 5;        // 0..3 : M strip
    const int mrow = lane & 15;
    const int half = lane >> 4;
    const int m0 = blockIdx.x * TILE_M;
    const int n0 = blockIdx.y * TILE_N;

    // async-copy assignment: 4 threads per 32-half row (16B segments),
    // 128 threads cover 32 rows per issue; 2 issues cover 64 rows.
    const int crow = t >> 2;              // 0..31
    const int cseg = (t & 3) * 8;         // half-element offset of 16B segment

    const uint32_t ldsA = (uint32_t)(size_t)(&sA[0]);
    const uint32_t ldsW = (uint32_t)(size_t)(&sW[0]);

    v8f z = {0.f, 0.f, 0.f, 0.f, 0.f, 0.f, 0.f, 0.f};
    v8f acc[4] = {z, z, z, z};

    for (int kk = 0; kk < K; kk += TILE_K) {
#pragma unroll
        for (int i = 0; i < 2; ++i) {
            const int r = crow + 32 * i;
            async_copy_b128(ldsA + (uint32_t)((r * TILE_K + cseg) * 2),
                            (uint64_t)(size_t)(A + (size_t)(m0 + r) * K + kk + cseg));
            async_copy_b128(ldsW + (uint32_t)((r * TILE_K + cseg) * 2),
                            (uint64_t)(size_t)(W + (size_t)(n0 + r) * K + kk + cseg));
        }
        wait_async0();          // this wave's async copies landed in LDS
        __syncthreads();        // all waves' copies landed

        Frag a;
        const int ml = wave * 16 + mrow;
#pragma unroll
        for (int v = 0; v < 8; ++v) {
            int k = ((v >> 2) << 4) + (half << 3) + ((v & 3) << 1);
            a.u[v] = *(const uint32_t*)&sA[ml * TILE_K + k];
        }
#pragma unroll
        for (int g = 0; g < 4; ++g) {
            Frag b;  // lane%16 = n_local within group g, elem j = k_local (table order)
#pragma unroll
            for (int j = 0; j < 16; ++j) b.h[j] = sW[(g * 16 + mrow) * TILE_K + ((j >> 3) << 4) + (half << 3) + (j & 7)];
            acc[g] = WMMA_F16(a.v, b.v, acc[g]);
        }
        __syncthreads();        // LDS reads done before next k-step overwrites
    }

    // C layout: lane(0-15)=N, VGPR v = M (lanes 16-31: M+8)
    const int rbase = m0 + wave * 16 + (half << 3);
#pragma unroll
    for (int g = 0; g < 4; ++g) {
        const int n = n0 + g * 16 + mrow;
#pragma unroll
        for (int v = 0; v < 8; ++v) {
            C[(size_t)(rbase + v) * N + n] = (OUT)acc[g][v];
        }
    }
}

// ---------------------------------------------------------------------------
// Flash attention: one wave per (b, h, 16-query tile), full (non-causal) attn.
// Q,K,V,O in f16 with layout [B*S, H*dk] (dk = 64).
// ---------------------------------------------------------------------------
#define ATT_S  2048
#define ATT_D  1024
#define ATT_H  16
#define ATT_DK 64

__global__ __launch_bounds__(32) void attn_kernel(const _Float16* __restrict__ Qh,
                                                  const _Float16* __restrict__ Kh,
                                                  const _Float16* __restrict__ Vh,
                                                  _Float16* __restrict__ Oh) {
    __shared__ _Float16 sP[16 * 32];  // P tile staged C-layout -> A-layout

    const int lane = threadIdx.x;
    const int mrow = lane & 15;
    const int half = lane >> 4;
    const int qt = blockIdx.x & 127;          // S/16 query tiles
    const int h  = (blockIdx.x >> 7) & 15;
    const int b  = blockIdx.x >> 11;
    const int q0 = qt * 16;

    // --- load Q A-fragments for K-dim 0..31 and 32..63 ---
    Frag aq0, aq1;
    const _Float16* __restrict__ qbase =
        Qh + (size_t)(b * ATT_S + q0 + mrow) * ATT_D + h * ATT_DK;
#pragma unroll
    for (int v = 0; v < 8; ++v) {
        int k = ((v >> 2) << 4) + (half << 3) + ((v & 3) << 1);
        aq0.u[v] = *(const uint32_t*)(qbase + k);
        aq1.u[v] = *(const uint32_t*)(qbase + 32 + k);
    }

    v8f z = {0.f, 0.f, 0.f, 0.f, 0.f, 0.f, 0.f, 0.f};
    v8f acc[4] = {z, z, z, z};
    float m_run[8], l_run[8];
#pragma unroll
    for (int v = 0; v < 8; ++v) { m_run[v] = -1e30f; l_run[v] = 0.f; }

    for (int base = 0; base < ATT_S; base += 32) {
        // ---- scores for 32 keys (two 16-key subtiles) ----
        v8f s0 = z, s1 = z;
#pragma unroll
        for (int t = 0; t < 2; ++t) {
#pragma unroll
            for (int f = 0; f < 2; ++f) {
                Frag kb;  // B-frag: lane%16 = key, elem = d (K-dim, table order)
                const int d = f * 32;
                const _Float16* __restrict__ kp =
                    Kh + (size_t)(b * ATT_S + base + t * 16 + mrow) * ATT_D + h * ATT_DK + d;
                __builtin_prefetch(kp + 32 * ATT_D, 0, 0);
#pragma unroll
                for (int j = 0; j < 16; ++j) kb.h[j] = kp[((j >> 3) << 4) + (half << 3) + (j & 7)];
                if (t == 0) s0 = WMMA_F16((f == 0) ? aq0.v : aq1.v, kb.v, s0);
                else        s1 = WMMA_F16((f == 0) ? aq0.v : aq1.v, kb.v, s1);
            }
        }

        // ---- online softmax: rows live per-VGPR across each 16-lane half ----
#pragma unroll
        for (int v = 0; v < 8; ++v) {
            float x0 = s0[v] * 0.125f;  // 1/sqrt(64)
            float x1 = s1[v] * 0.125f;
            float mx = fmaxf(x0, x1);
#pragma unroll
            for (int msk = 1; msk < 16; msk <<= 1)
                mx = fmaxf(mx, __shfl_xor(mx, msk, 32));
            float nm = fmaxf(m_run[v], mx);
            float p0 = __expf(x0 - nm);
            float p1 = __expf(x1 - nm);
            float rs = p0 + p1;
#pragma unroll
            for (int msk = 1; msk < 16; msk <<= 1)
                rs += __shfl_xor(rs, msk, 32);
            float al = __expf(m_run[v] - nm);
            l_run[v] = l_run[v] * al + rs;
            m_run[v] = nm;
            acc[0][v] *= al; acc[1][v] *= al; acc[2][v] *= al; acc[3][v] *= al;
            // P tile to LDS (row = query, col = key-within-32)
            const int r = (half << 3) + v;
            sP[r * 32 + mrow]      = (_Float16)p0;
            sP[r * 32 + mrow + 16] = (_Float16)p1;
        }
        __syncthreads();

        // ---- re-load P as A-fragment (lane = query row, elems = key) ----
        Frag pa;
#pragma unroll
        for (int v = 0; v < 8; ++v) {
            int k = ((v >> 2) << 4) + (half << 3) + ((v & 3) << 1);
            pa.u[v] = *(const uint32_t*)&sP[mrow * 32 + k];
        }
        __syncthreads();

        // ---- O += P @ V  (4 output groups of 16 along dk) ----
#pragma unroll
        for (int g = 0; g < 4; ++g) {
            Frag vb;  // lane%16 = dk within group, elems = key-within-32 (table order)
            const _Float16* __restrict__ vp =
                Vh + (size_t)(b * ATT_S + base) * ATT_D + h * ATT_DK + g * 16 + mrow;
#pragma unroll
            for (int j = 0; j < 16; ++j) vb.h[j] = vp[(size_t)(((j >> 3) << 4) + (half << 3) + (j & 7)) * ATT_D];
            acc[g] = WMMA_F16(pa.v, vb.v, acc[g]);
        }
    }

    // ---- epilogue: normalize by l and store ----
#pragma unroll
    for (int g = 0; g < 4; ++g) {
#pragma unroll
        for (int v = 0; v < 8; ++v) {
            const int row = q0 + (half << 3) + v;
            const int d = h * ATT_DK + g * 16 + mrow;
            Oh[(size_t)(b * ATT_S + row) * ATT_D + d] =
                (_Float16)(acc[g][v] * (1.0f / l_run[v]));
        }
    }
}

// ---------------------------------------------------------------------------
// out = LayerNorm(x + y) * g + b     (one 256-thread block per token, D=1024)
// ---------------------------------------------------------------------------
__global__ __launch_bounds__(256) void add_ln_kernel(const float* __restrict__ X,
                                                     const float* __restrict__ Y,
                                                     const float* __restrict__ gam,
                                                     const float* __restrict__ bet,
                                                     float* __restrict__ out) {
    __shared__ float rs[256], rq[256];
    const int t = threadIdx.x;
    const size_t row = blockIdx.x;
    const float* __restrict__ x = X + row * 1024;
    const float* __restrict__ y = Y + row * 1024;

    float vbuf[4], s = 0.f, sq = 0.f;
#pragma unroll
    for (int r = 0; r < 4; ++r) {
        int d = t + 256 * r;
        vbuf[r] = x[d] + y[d];
        s += vbuf[r];
        sq += vbuf[r] * vbuf[r];
    }
    rs[t] = s; rq[t] = sq;
    __syncthreads();
    for (int ofs = 128; ofs > 0; ofs >>= 1) {
        if (t < ofs) { rs[t] += rs[t + ofs]; rq[t] += rq[t + ofs]; }
        __syncthreads();
    }
    const float mean = rs[0] * (1.0f / 1024.0f);
    const float var  = rq[0] * (1.0f / 1024.0f) - mean * mean;
    const float rstd = rsqrtf(var + 1e-5f);
#pragma unroll
    for (int r = 0; r < 4; ++r) {
        int d = t + 256 * r;
        out[row * 1024 + d] = (vbuf[r] - mean) * rstd * gam[d] + bet[d];
    }
}

// ---------------------------------------------------------------------------
// Quantum FFN: one 256-thread block per token.
// ang = x1 @ Win^T + b_in ; simulate 8-qubit RX/RY/CNOT circuit on 256-amp
// statevector in LDS ; expz = Pauli-Z expectations ; out = relu(expz @ Wout^T + b_out)
// ---------------------------------------------------------------------------
__global__ __launch_bounds__(256) void qffn_kernel(const float* __restrict__ X1,
                                                   const float* __restrict__ Win,
                                                   const float* __restrict__ b_in,
                                                   const float* __restrict__ Wout,
                                                   const float* __restrict__ b_out,
                                                   const float* __restrict__ ry,
                                                   float* __restrict__ F) {
    __shared__ float sre[256], sim_[256];
    __shared__ float ang[8], expz[8];

    const int t = threadIdx.x;
    const int lane = t & 31;
    const int w = t >> 5;  // wave id 0..7 == qubit id
    const size_t tok = blockIdx.x;
    const float* __restrict__ x = X1 + tok * 1024;

    // angles: wave w computes ang[w] = <x1, Win[w]> + b_in[w]
    float a = 0.f;
#pragma unroll 4
    for (int i = 0; i < 32; ++i) {
        int d = lane + 32 * i;
        a += x[d] * Win[w * 1024 + d];
    }
#pragma unroll
    for (int msk = 1; msk < 32; msk <<= 1) a += __shfl_xor(a, msk, 32);
    if (lane == 0) ang[w] = a + b_in[w];

    sre[t] = (t == 0) ? 1.0f : 0.0f;
    sim_[t] = 0.0f;
    __syncthreads();

    // RX(ang[wi]) on wire wi (wire 0 = MSB)
    for (int wi = 0; wi < 8; ++wi) {
        const float th = 0.5f * ang[wi];
        const float c = __cosf(th), s = __sinf(th);
        const int shift = 7 - wi, stride = 1 << shift;
        if (t < 128) {
            int i0 = ((t >> shift) << (shift + 1)) | (t & (stride - 1));
            int i1 = i0 + stride;
            float ar = sre[i0], ai = sim_[i0], br = sre[i1], bi = sim_[i1];
            sre[i0] = c * ar + s * bi;   sim_[i0] = c * ai - s * br;
            sre[i1] = s * ai + c * br;   sim_[i1] = -s * ar + c * bi;
        }
        __syncthreads();
    }
    // RY(ry[wi]) on wire wi
    for (int wi = 0; wi < 8; ++wi) {
        const float th = 0.5f * ry[wi];
        const float c = __cosf(th), s = __sinf(th);
        const int shift = 7 - wi, stride = 1 << shift;
        if (t < 128) {
            int i0 = ((t >> shift) << (shift + 1)) | (t & (stride - 1));
            int i1 = i0 + stride;
            float ar = sre[i0], ai = sim_[i0], br = sre[i1], bi = sim_[i1];
            sre[i0] = c * ar - s * br;   sim_[i0] = c * ai - s * bi;
            sre[i1] = s * ar + c * br;   sim_[i1] = s * ai + c * bi;
        }
        __syncthreads();
    }
    // CNOT chain: control cw, target cw+1 (swap target bit where control==1)
    for (int cw = 0; cw < 7; ++cw) {
        const int bc = 7 - cw, bt = bc - 1;
        if (t < 64) {
            int lowmask = (1 << bt) - 1;
            int i0 = (t & lowmask) | (1 << bc) | ((t >> bt) << (bc + 1));
            int i1 = i0 | (1 << bt);
            float r0 = sre[i0], m0 = sim_[i0];
            sre[i0] = sre[i1]; sim_[i0] = sim_[i1];
            sre[i1] = r0;      sim_[i1] = m0;
        }
        __syncthreads();
    }

    // probabilities (reuse sre as prob buffer)
    float p = sre[t] * sre[t] + sim_[t] * sim_[t];
    sre[t] = p;
    __syncthreads();

    // expz[w] = sum_j probs[j] * (1 - 2*bit_{7-w}(j))   (wave w owns qubit w)
    float e = 0.f;
#pragma unroll
    for (int i = 0; i < 8; ++i) {
        int j = lane + 32 * i;
        e += ((j >> (7 - w)) & 1) ? -sre[j] : sre[j];
    }
#pragma unroll
    for (int msk = 1; msk < 32; msk <<= 1) e += __shfl_xor(e, msk, 32);
    if (lane == 0) expz[w] = e;
    __syncthreads();

    // out = relu(expz @ Wout^T + b_out)
#pragma unroll
    for (int r = 0; r < 4; ++r) {
        int d = t + 256 * r;
        float o = b_out[d];
#pragma unroll
        for (int q = 0; q < 8; ++q) o += expz[q] * Wout[d * 8 + q];
        F[tok * 1024 + d] = fmaxf(o, 0.0f);
    }
}

// ---------------------------------------------------------------------------
// Host-side orchestration
// ---------------------------------------------------------------------------
extern "C" void kernel_launch(void* const* d_in, const int* in_sizes, int n_in,
                              void* d_out, int out_size, void* d_ws, size_t ws_size,
                              hipStream_t stream) {
    (void)in_sizes; (void)n_in; (void)out_size; (void)ws_size;

    const float* x     = (const float*)d_in[0];
    const float* Wq    = (const float*)d_in[1];
    const float* Wk    = (const float*)d_in[2];
    const float* Wv    = (const float*)d_in[3];
    const float* Wo    = (const float*)d_in[4];
    const float* g1    = (const float*)d_in[5];
    const float* b1    = (const float*)d_in[6];
    const float* g2    = (const float*)d_in[7];
    const float* b2    = (const float*)d_in[8];
    const float* Win   = (const float*)d_in[9];
    const float* b_in  = (const float*)d_in[10];
    const float* Wout  = (const float*)d_in[11];
    const float* b_out = (const float*)d_in[12];
    const float* ry    = (const float*)d_in[13];
    float* out = (float*)d_out;

    const int Mtok = 4096;   // B*S
    const int D = 1024;

    size_t off = 0;
    char* wsb = (char*)d_ws;
    auto take = [&](size_t nbytes) -> char* {
        char* p = wsb + off;
        off += (nbytes + 255) & ~(size_t)255;
        return p;
    };
    _Float16* Xh  = (_Float16*)take((size_t)Mtok * D * 2);
    _Float16* Wqh = (_Float16*)take((size_t)D * D * 2);
    _Float16* Wkh = (_Float16*)take((size_t)D * D * 2);
    _Float16* Wvh = (_Float16*)take((size_t)D * D * 2);
    _Float16* Woh = (_Float16*)take((size_t)D * D * 2);
    _Float16* Qh  = (_Float16*)take((size_t)Mtok * D * 2);
    _Float16* Kh  = (_Float16*)take((size_t)Mtok * D * 2);
    _Float16* Vh  = (_Float16*)take((size_t)Mtok * D * 2);
    _Float16* Oh  = (_Float16*)take((size_t)Mtok * D * 2);
    float* attn_o = (float*)take((size_t)Mtok * D * 4);
    float* x1     = (float*)take((size_t)Mtok * D * 4);
    float* ffn    = (float*)take((size_t)Mtok * D * 4);

    // 1) f32 -> f16 conversions
    cvt_f16_kernel<<<(Mtok * D + 255) / 256, 256, 0, stream>>>(x, Xh, Mtok * D);
    cvt_f16_kernel<<<(D * D + 255) / 256, 256, 0, stream>>>(Wq, Wqh, D * D);
    cvt_f16_kernel<<<(D * D + 255) / 256, 256, 0, stream>>>(Wk, Wkh, D * D);
    cvt_f16_kernel<<<(D * D + 255) / 256, 256, 0, stream>>>(Wv, Wvh, D * D);
    cvt_f16_kernel<<<(D * D + 255) / 256, 256, 0, stream>>>(Wo, Woh, D * D);

    // 2) Q/K/V projections (WMMA + async-LDS staging, f16 out)
    dim3 ggrid(Mtok / TILE_M, D / TILE_N);
    gemm_kernel<_Float16><<<ggrid, 128, 0, stream>>>(Xh, Wqh, Qh, Mtok, D, D);
    gemm_kernel<_Float16><<<ggrid, 128, 0, stream>>>(Xh, Wkh, Kh, Mtok, D, D);
    gemm_kernel<_Float16><<<ggrid, 128, 0, stream>>>(Xh, Wvh, Vh, Mtok, D, D);

    // 3) attention (WMMA flash-attention), O in f16
    attn_kernel<<<2 * 16 * (2048 / 16), 32, 0, stream>>>(Qh, Kh, Vh, Oh);

    // 4) output projection (WMMA, f32 out)
    gemm_kernel<float><<<ggrid, 128, 0, stream>>>(Oh, Woh, attn_o, Mtok, D, D);

    // 5) x1 = LN(x + attn_out)
    add_ln_kernel<<<Mtok, 256, 0, stream>>>(x, attn_o, g1, b1, x1);

    // 6) quantum FFN
    qffn_kernel<<<Mtok, 256, 0, stream>>>(x1, Win, b_in, Wout, b_out, ry, ffn);

    // 7) out = LN(x1 + ffn)
    add_ln_kernel<<<Mtok, 256, 0, stream>>>(x1, ffn, g2, b2, out);
}
